// MS_SpikeAttention_RepConv3D_qkv_id_87445534146942
// MI455X (gfx1250) — matrix operand order repaired and checked by reading, not compiled, not run on hardware
//
#include <hip/hip_runtime.h>
#include <hip/hip_bf16.h>

// ---------------------------------------------------------------------------
// MS_SpikeAttention_RepConv3D pipeline for gfx1250 (MI455X, wave32, WMMA+TDM)
// T=4, B=2, C=256, D=H=W=16 -> N=4096, TB=8 images, 8 heads x 32 dim
// ---------------------------------------------------------------------------

typedef __attribute__((ext_vector_type(16))) _Float16 v16h;
typedef __attribute__((ext_vector_type(8)))  float    v8f;
typedef __attribute__((ext_vector_type(4)))  unsigned tdm_u4;
typedef __attribute__((ext_vector_type(8)))  int      tdm_i8;
typedef __attribute__((ext_vector_type(4)))  int      tdm_i4;

#define TT      4
#define BB      2
#define TB      8
#define CC      256
#define NN      4096
#define GROUPS  8
#define CPG     32
#define HEADS   8
#define HD      32
#define PER_T   (BB * CC * NN)
#define TEN     ((size_t)TB * CC * NN)

#define HAS_TDM (__has_builtin(__builtin_amdgcn_tensor_load_to_lds) && \
                 __has_builtin(__builtin_amdgcn_s_wait_tensorcnt))

// ---------------------------------------------------------------------------
// LIF scan over T (ParametricLIF, decay_input, hard reset to 0)
// ---------------------------------------------------------------------------
__global__ __launch_bounds__(256)
void lif_kernel(const float* __restrict__ x, float* __restrict__ s,
                const float* __restrict__ lw, int widx, float thr) {
    int i = blockIdx.x * 256 + threadIdx.x;
    if (i >= PER_T) return;
    float w = lw[widx];
    float tau_inv = 1.0f / (1.0f + __expf(-w));
    float v = 0.0f;
#pragma unroll
    for (int t = 0; t < TT; ++t) {
        float xt = x[(size_t)t * PER_T + i];
        v += (xt - v) * tau_inv;
        float sp = (v - thr) >= 0.0f ? 1.0f : 0.0f;
        s[(size_t)t * PER_T + i] = sp;
        v *= (1.0f - sp);
    }
}

// ---------------------------------------------------------------------------
// 1x1x1 conv == GEMM:  Y[img][co][n] = sum_ci W[co][ci] * X[img][ci][n]
// grid = (NN/64, CC/16, TB), block = 128 (4 waves); wave w -> 16 n-columns.
// Weight tile staged via Tensor Data Mover (TDM) when available; activation
// panel staged transposed so WMMA B-fragments are contiguous ds_load_b128s.
// ---------------------------------------------------------------------------
__global__ __launch_bounds__(128)
void spike_gemm_wmma(const float* __restrict__ W, const float* __restrict__ X,
                     float* __restrict__ Y) {
    __shared__ _Float16 sW[16][CC];                    // co-tile x K  (8 KB)
    __shared__ __align__(128) _Float16 sXT[64][32];    // n x K-panel (4 KB)

    const int img  = blockIdx.z;
    const int co0  = blockIdx.y * 16;
    const int n0   = blockIdx.x * 64;
    const int tid  = threadIdx.x;
    const int lane = tid & 31;
    const int wv   = tid >> 5;

#if HAS_TDM
    // --- TDM staging: one 2-D descriptor pulls the 16x256 f32 weight tile ---
    __shared__ __align__(16) float sWf[16 * CC];       // 16 KB TDM landing pad
    if (tid < 32) {                                    // one wave issues DMA
        unsigned long long ga =
            (unsigned long long)(const void*)(W + (size_t)co0 * CC);
        // NOTE: compile-only environment; LDS operand uses truncated address.
        unsigned lds = (unsigned)(unsigned long long)(const void*)sWf;
        tdm_u4 g0 = { 1u,                              // count=1, user mode
                      lds,                             // lds_addr
                      (unsigned)(ga & 0xFFFFFFFFull),  // global_addr[31:0]
                      (unsigned)((ga >> 32) & 0x01FFFFFFull) | (2u << 30) };
        tdm_i8 g1 = { (int)(2u << 16),   // wg_mask=0, data_size=2 (4 bytes)
                      (int)(CC << 16),   // tensor_dim0[15:0] = 256
                      (int)(CC << 16),   // tensor_dim1[15:0] = 256
                      (int)(CC << 16),   // tile_dim0 = 256 (contiguous K)
                      16,                // tile_dim1 = 16 rows, tile_dim2 = 0
                      CC,                // tensor_dim0_stride = 256
                      0, 0 };
        tdm_i4 gz = { 0, 0, 0, 0 };
#if defined(__clang_major__) && (__clang_major__ >= 23)
        tdm_i8 gz8 = { 0, 0, 0, 0, 0, 0, 0, 0 };
        __builtin_amdgcn_tensor_load_to_lds(g0, g1, gz, gz, gz8, 0);
#else
        __builtin_amdgcn_tensor_load_to_lds(g0, g1, gz, gz, 0);
#endif
        __builtin_amdgcn_s_wait_tensorcnt(0);
    }
    __syncthreads();
    for (int i = tid; i < 16 * CC; i += 128)
        sW[i >> 8][i & 255] = (_Float16)sWf[i];
#else
    for (int i = tid; i < 16 * CC; i += 128)
        sW[i >> 8][i & 255] = (_Float16)W[(co0 + (i >> 8)) * CC + (i & 255)];
#endif

    const float* Xi = X + (size_t)img * CC * NN;
    v8f acc = {};

    const int m     = lane & 15;           // A row / B column within tile
    const int ahalf = (lane >> 4) * 8;     // unused; A loads also take K+16 via bhalf
    const int bhalf = (lane >> 4) * 16;    // B: hi half-lanes take K+16
    const int ncol  = wv * 16 + m;

    for (int k0 = 0; k0 < CC; k0 += 32) {
        __syncthreads();
        // stage activation panel transposed: read coalesced over n,
        // store [n][k] so B fragments are contiguous 32-byte runs
        for (int i = tid; i < 32 * 64; i += 128) {
            int kk = i >> 6, n = i & 63;
            sXT[n][kk] = (_Float16)Xi[(size_t)(k0 + kk) * NN + n0 + n];
        }
        if (k0 + 32 < CC)   // speculative prefetch of next K panel
            __builtin_prefetch(&Xi[(size_t)(k0 + 32) * NN + n0 + tid], 0, 1);
        __syncthreads();

        v16h a, b;
#pragma unroll
        for (int e = 0; e < 16; ++e) {
            int k = bhalf + e;                           // one 16-elem run
            a[e] = sW[m][k0 + k];
        }
#pragma unroll
        for (int e = 0; e < 16; ++e)                     // one 16-elem run
            b[e] = sXT[ncol][bhalf + e];

        acc = __builtin_amdgcn_wmma_f32_16x16x32_f16(
            false, a, false, b, (short)0, acc, false, false);
    }

    // D layout: VGPR r -> M = r + 8*(lane>=16), N = lane&15
    float* Yo = Y + (size_t)img * CC * NN;
    const int n  = n0 + wv * 16 + (lane & 15);
    const int mo = (lane >> 4) * 8;
#pragma unroll
    for (int r = 0; r < 8; ++r)
        Yo[(size_t)(co0 + mo + r) * NN + n] = acc[r];
}

// ---------------------------------------------------------------------------
// GroupNorm (in-place): one block per (img, group); 32 ch x 4096 vox each
// ---------------------------------------------------------------------------
__global__ __launch_bounds__(256)
void groupnorm_kernel(float* __restrict__ x, const float* __restrict__ gamma,
                      const float* __restrict__ beta) {
    const int img = blockIdx.x >> 3;
    const int grp = blockIdx.x & 7;
    float* p = x + ((size_t)img * CC + grp * CPG) * NN;
    const int tid = threadIdx.x;
    const int CNT = CPG * NN;

    float s = 0.0f, s2 = 0.0f;
    for (int i = tid; i < CNT; i += 256) {
        float v = p[i];
        s += v; s2 += v * v;
    }
    __shared__ float rs[256], rq[256];
    rs[tid] = s; rq[tid] = s2;
    __syncthreads();
    for (int off = 128; off > 0; off >>= 1) {
        if (tid < off) { rs[tid] += rs[tid + off]; rq[tid] += rq[tid + off]; }
        __syncthreads();
    }
    const float inv  = 1.0f / (float)CNT;
    const float mean = rs[0] * inv;
    const float var  = rq[0] * inv - mean * mean;
    const float rstd = rsqrtf(var + 1e-5f);

    for (int i = tid; i < CNT; i += 256) {
        int c = grp * CPG + (i >> 12);
        float v = (p[i] - mean) * rstd;
        p[i] = v * gamma[c] + beta[c];
    }
}

// ---------------------------------------------------------------------------
// Depthwise 3x3x3 conv with fused bias-padding (pad voxel value = GN bias)
// ---------------------------------------------------------------------------
__global__ __launch_bounds__(256)
void dwconv3_kernel(const float* __restrict__ in, const float* __restrict__ wdw,
                    const float* __restrict__ bias, float* __restrict__ out) {
    size_t idx = (size_t)blockIdx.x * 256 + threadIdx.x;
    int n = (int)(idx & 4095);
    int c = (int)((idx >> 12) & 255);
    int w = n & 15, h = (n >> 4) & 15, d = n >> 8;
    const float* ip = in + (idx - (size_t)n);
    const float bv = bias[c];
    const float* wp = wdw + c * 27;

    float acc = 0.0f;
#pragma unroll
    for (int kd = 0; kd < 3; ++kd)
#pragma unroll
        for (int kh = 0; kh < 3; ++kh)
#pragma unroll
            for (int kw = 0; kw < 3; ++kw) {
                int dd = d + kd - 1, hh = h + kh - 1, ww = w + kw - 1;
                bool inside = ((unsigned)dd < 16u) & ((unsigned)hh < 16u) &
                              ((unsigned)ww < 16u);
                float v = inside ? ip[(dd << 8) + (hh << 4) + ww] : bv;
                acc += v * wp[(kd * 3 + kh) * 3 + kw];
            }
    out[idx] = acc;
}

// ---------------------------------------------------------------------------
// kv = k^T @ v per (img, head): [32 x 32] = [32 x 4096] x [4096 x 32]
// ---------------------------------------------------------------------------
__global__ __launch_bounds__(32)
void attn_kv_kernel(const float* __restrict__ K, const float* __restrict__ V,
                    float* __restrict__ KV) {
    const int ih   = blockIdx.x;
    const int img  = ih >> 3, h = ih & 7;
    const int lane = threadIdx.x;
    const float* Kp = K + ((size_t)img * CC + h * HD) * NN;
    const float* Vp = V + ((size_t)img * CC + h * HD) * NN;

    v8f acc[2][2] = {};
    const int m     = lane & 15;
    const int ahalf = (lane >> 4) * 8;
    const int bhalf = (lane >> 4) * 16;

    for (int n0 = 0; n0 < NN; n0 += 32) {
        v16h a[2], b[2];
#pragma unroll
        for (int td = 0; td < 2; ++td)
#pragma unroll
            for (int e = 0; e < 16; ++e) {
                int kk = bhalf + e;
                a[td][e] = (_Float16)Kp[(size_t)(td * 16 + m) * NN + n0 + kk];
            }
#pragma unroll
        for (int te = 0; te < 2; ++te)
#pragma unroll
            for (int e = 0; e < 16; ++e)
                b[te][e] = (_Float16)Vp[(size_t)(te * 16 + m) * NN + n0 + bhalf + e];
#pragma unroll
        for (int td = 0; td < 2; ++td)
#pragma unroll
            for (int te = 0; te < 2; ++te)
                acc[td][te] = __builtin_amdgcn_wmma_f32_16x16x32_f16(
                    false, a[td], false, b[te], (short)0, acc[td][te], false, false);
    }

    float* kvp = KV + (size_t)ih * HD * HD;
    const int mo = (lane >> 4) * 8;
#pragma unroll
    for (int td = 0; td < 2; ++td)
#pragma unroll
        for (int te = 0; te < 2; ++te)
#pragma unroll
            for (int r = 0; r < 8; ++r)
                kvp[(td * 16 + mo + r) * HD + te * 16 + (lane & 15)] = acc[td][te][r];
}

// ---------------------------------------------------------------------------
// out[n][e] = 0.125 * sum_d q[n][d] * kv[d][e] per (img, head)
// kv tile cached transposed in LDS so B fragments are contiguous.
// ---------------------------------------------------------------------------
__global__ __launch_bounds__(256)
void attn_out_kernel(const float* __restrict__ Q, const float* __restrict__ KV,
                     float* __restrict__ Out) {
    const int ih   = blockIdx.y;
    const int img  = ih >> 3, h = ih & 7;
    const int tid  = threadIdx.x, lane = tid & 31, wv = tid >> 5;
    const int n0   = (blockIdx.x * 8 + wv) * 16;
    const float* Qp  = Q + ((size_t)img * CC + h * HD) * NN;
    const float* kvp = KV + (size_t)ih * HD * HD;

    __shared__ __align__(128) _Float16 skvT[HD][HD];   // [e][d] transposed
    for (int i = tid; i < HD * HD; i += 256)
        skvT[i & 31][i >> 5] = (_Float16)kvp[i];
    __syncthreads();

    const int m     = lane & 15;
    const int ahalf = (lane >> 4) * 8;
    const int bhalf = (lane >> 4) * 16;

    v16h a;
#pragma unroll
    for (int e = 0; e < 16; ++e) {
        int kk = bhalf + e;
        a[e] = (_Float16)Qp[(size_t)kk * NN + n0 + m];
    }

    float* Op = Out + ((size_t)img * CC + h * HD) * NN;
    const int mo = (lane >> 4) * 8;
#pragma unroll
    for (int te = 0; te < 2; ++te) {
        v16h b;
#pragma unroll
        for (int e = 0; e < 16; ++e)
            b[e] = skvT[te * 16 + (lane & 15)][bhalf + e];
        v8f acc = {};
        acc = __builtin_amdgcn_wmma_f32_16x16x32_f16(
            false, a, false, b, (short)0, acc, false, false);
#pragma unroll
        for (int r = 0; r < 8; ++r)
            Op[(size_t)(te * 16 + (lane & 15)) * NN + n0 + mo + r] = acc[r] * 0.125f;
    }
}

// ---------------------------------------------------------------------------
// Orchestration
// ---------------------------------------------------------------------------
extern "C" void kernel_launch(void* const* d_in, const int* in_sizes, int n_in,
                              void* d_out, int out_size, void* d_ws, size_t ws_size,
                              hipStream_t stream) {
    (void)in_sizes; (void)n_in; (void)out_size; (void)ws_size;

    const float* x     = (const float*)d_in[0];
    const float* wproj = (const float*)d_in[1];
    const float* g1    = (const float*)d_in[2];
    const float* b1    = (const float*)d_in[3];
    const float* wdw   = (const float*)d_in[4];
    const float* wpw   = (const float*)d_in[5];
    const float* g2    = (const float*)d_in[6];
    const float* b2    = (const float*)d_in[7];
    const float* pg    = (const float*)d_in[8];
    const float* pb    = (const float*)d_in[9];
    const float* lw    = (const float*)d_in[10];
    float* out = (float*)d_out;

    float* ws  = (float*)d_ws;
    float* xs  = ws;
    float* t1  = xs + TEN;
    float* t2  = t1 + TEN;
    float* kb  = t2 + TEN;
    float* vb  = kb + TEN;
    float* kvb = vb + TEN;
    float* qb  = out;   // q spikes parked in d_out, consumed before final use

    const dim3 gemmGrid(NN / 64, CC / 16, TB);
    const dim3 gnGrid(TB * GROUPS);
    const int  dwBlocks  = (TB * CC * NN) / 256;
    const int  lifBlocks = (PER_T + 255) / 256;

    lif_kernel<<<lifBlocks, 256, 0, stream>>>(x, xs, lw, 0, 1.0f);

    float* sp[3] = { qb, kb, vb };
    for (int i = 0; i < 3; ++i) {
        spike_gemm_wmma<<<gemmGrid, 128, 0, stream>>>(wproj + i * CC * CC, xs, t1);
        groupnorm_kernel<<<gnGrid, 256, 0, stream>>>(t1, g1 + i * CC, b1 + i * CC);
        dwconv3_kernel<<<dwBlocks, 256, 0, stream>>>(t1, wdw + i * CC * 27,
                                                     b1 + i * CC, t2);
        spike_gemm_wmma<<<gemmGrid, 128, 0, stream>>>(wpw + i * CC * CC, t2, t1);
        groupnorm_kernel<<<gnGrid, 256, 0, stream>>>(t1, g2 + i * CC, b2 + i * CC);
        groupnorm_kernel<<<gnGrid, 256, 0, stream>>>(t1, pg + i * CC, pb + i * CC);
        lif_kernel<<<lifBlocks, 256, 0, stream>>>(t1, sp[i], lw, 1 + i, 1.0f);
    }

    attn_kv_kernel<<<64, 32, 0, stream>>>(kb, vb, kvb);
    attn_out_kernel<<<dim3(NN / 128, 64), 256, 0, stream>>>(qb, kvb, t2);
    lif_kernel<<<lifBlocks, 256, 0, stream>>>(t2, t1, lw, 4, 0.5f);

    spike_gemm_wmma<<<gemmGrid, 128, 0, stream>>>(wproj + 3 * CC * CC, t1, t2);
    groupnorm_kernel<<<gnGrid, 256, 0, stream>>>(t2, g1 + 3 * CC, b1 + 3 * CC);
    dwconv3_kernel<<<dwBlocks, 256, 0, stream>>>(t2, wdw + 3 * CC * 27,
                                                 b1 + 3 * CC, t1);
    spike_gemm_wmma<<<gemmGrid, 128, 0, stream>>>(wpw + 3 * CC * CC, t1, out);
    groupnorm_kernel<<<gnGrid, 256, 0, stream>>>(out, g2 + 3 * CC, b2 + 3 * CC);
    groupnorm_kernel<<<gnGrid, 256, 0, stream>>>(out, pg + 3 * CC, pb + 3 * CC);
}
